// GatedInternalEdgeCompletionHead_19344532702120
// MI455X (gfx1250) — matrix operand order repaired and checked by reading, not compiled, not run on hardware
//
#include <hip/hip_runtime.h>
#include <hip/hip_bf16.h>

typedef _Float16 h2   __attribute__((ext_vector_type(2)));
typedef _Float16 v16h __attribute__((ext_vector_type(16)));
typedef float    v8f  __attribute__((ext_vector_type(8)));

#define BDIM 4
#define NDIM 256
#define HDIM 256
#define HID  256
#define NT8  32          // number of 8-wide tiles along N
#define NTRI 528         // NT8*(NT8+1)/2

// ---------------------------------------------------------------------------
// Kernel 1a: pre1[b,i,k] = b1[k] + sum_h node[b,i,h]*W1[h,k]
//            pre2[b,i,k] =         sum_h node[b,i,h]*W1[256+h,k]
// ---------------------------------------------------------------------------
__global__ __launch_bounds__(256)
void precompute_kernel(const float* __restrict__ node,
                       const float* __restrict__ W1,
                       const float* __restrict__ b1,
                       float* __restrict__ pre1,
                       float* __restrict__ pre2) {
  const int bi = blockIdx.x;            // b*256 + i
  const int k  = threadIdx.x;           // hidden unit
  __shared__ float row[HDIM];
  row[k] = node[bi * HDIM + k];
  __syncthreads();
  float a1 = b1[k];
  float a2 = 0.0f;
#pragma unroll 8
  for (int h = 0; h < HDIM; ++h) {
    const float r = row[h];
    a1 = fmaf(r, W1[h * HID + k], a1);
    a2 = fmaf(r, W1[(HDIM + h) * HID + k], a2);
  }
  pre1[bi * HID + k] = a1;
  pre2[bi * HID + k] = a2;
}

// ---------------------------------------------------------------------------
// Kernel 1b: w1T[k][f] = (f16) W1[512+f][k], f in [0,512): first 256 = |diff|
//            weights, next 256 = product weights.  Layout [HID=256][512].
// ---------------------------------------------------------------------------
__global__ __launch_bounds__(256)
void w1t_kernel(const float* __restrict__ W1, _Float16* __restrict__ w1T) {
  const int f = blockIdx.x;             // 0..511
  const int k = threadIdx.x;            // 0..255  (coalesced read)
  w1T[k * 512 + f] = (_Float16)W1[(512 + f) * HID + k];
}

// ---------------------------------------------------------------------------
// Kernel 2: symmetric pairwise GEMM (WMMA f16->f32), upper-triangular tiles,
// dual epilogue producing raw[i][j] and raw[j][i].
// Workgroup = 256 threads = 8 wave32; 8 i x 8 j = 64 pairs (M=64, K=512, N=256).
// Wave w = (mt = w&3 -> M-tile, nh = w>>2 -> N half); 8 acc tiles per wave.
// ---------------------------------------------------------------------------
__global__ __launch_bounds__(256)
void edge_mlp_main(const float* __restrict__ node,
                   const float* __restrict__ logits,
                   const float* __restrict__ W1,
                   const float* __restrict__ W2,
                   const float* __restrict__ b2,
                   const float* __restrict__ pre1,
                   const float* __restrict__ pre2,
                   const _Float16* __restrict__ w1T,
                   float* __restrict__ raw) {
  // decode upper-triangular tile pair (it <= jt)
  int rem = blockIdx.x, it = 0;
  while (rem >= (NT8 - it)) { rem -= (NT8 - it); ++it; }
  const int jt = it + rem;
  const int b  = blockIdx.y;
  const int i0 = it * 8;
  const int j0 = jt * 8;

  const int t    = threadIdx.x;
  const int lane = t & 31;
  const int wave = t >> 5;
  const int mt   = wave & 3;            // M-tile (16 pairs)
  const int nh   = wave >> 2;           // N half (8 N-tiles)

  __shared__ __align__(16) char s_u[48 * 1024];
  float*    s_nodeI = reinterpret_cast<float*>(s_u);                 //  8 KB
  float*    s_nodeJ = reinterpret_cast<float*>(s_u + 8 * 1024);      //  8 KB
  _Float16* s_feat  = reinterpret_cast<_Float16*>(s_u + 16 * 1024);  //  4 KB
  _Float16* s_w1T   = reinterpret_cast<_Float16*>(s_u + 20 * 1024);  // 16 KB
  // epilogue overlays
  float* s_p1I = reinterpret_cast<float*>(s_u);                      //  8 KB
  float* s_p2J = reinterpret_cast<float*>(s_u + 8 * 1024);           //  8 KB
  float* s_p1J = reinterpret_cast<float*>(s_u + 16 * 1024);          //  8 KB
  float* s_p2I = reinterpret_cast<float*>(s_u + 24 * 1024);          //  8 KB
  float* s_red = reinterpret_cast<float*>(s_u + 32 * 1024);          // 16 KB

  __shared__ float s_w1e[256], s_w1f[256], s_w2[256];
  __shared__ float s_clvF[64], s_sgvF[64], s_clvR[64], s_sgvR[64];

  const float b2v = b2[0];

  // -------- one-time staging: node tiles, gate scalars, small vectors ------
  {
    const int row = t >> 5, cb = (t & 31) * 8;
    const float4* gi =
        reinterpret_cast<const float4*>(node + ((b * NDIM + i0 + row) * HDIM + cb));
    const float4* gj =
        reinterpret_cast<const float4*>(node + ((b * NDIM + j0 + row) * HDIM + cb));
    float4* di = reinterpret_cast<float4*>(&s_nodeI[row * 256 + cb]);
    float4* dj = reinterpret_cast<float4*>(&s_nodeJ[row * 256 + cb]);
    di[0] = gi[0]; di[1] = gi[1];
    dj[0] = gj[0]; dj[1] = gj[1];
  }
  if (t < 64) {
    const int p = t;
    const int i = i0 + (p >> 3), j = j0 + (p & 7);
    float cf = logits[(b * NDIM + i) * NDIM + j];
    float cr = logits[(b * NDIM + j) * NDIM + i];
    cf = fminf(fmaxf(cf, -20.0f), 20.0f);
    cr = fminf(fmaxf(cr, -20.0f), 20.0f);
    s_clvF[p] = cf; s_sgvF[p] = 1.0f / (1.0f + __expf(-cf));
    s_clvR[p] = cr; s_sgvR[p] = 1.0f / (1.0f + __expf(-cr));
  }
  s_w1e[t] = W1[1024 * HID + t];
  s_w1f[t] = W1[1025 * HID + t];
  s_w2[t]  = W2[t];

  // -------- accumulators: 8 N-tiles for this wave's (mt, nh) ---------------
  v8f acc[8];
  const v8f zero = {0.f, 0.f, 0.f, 0.f, 0.f, 0.f, 0.f, 0.f};
#pragma unroll
  for (int n = 0; n < 8; ++n) acc[n] = zero;

  const int p_me = t >> 2;              // feat-staging pair (4 threads / pair)
  const int iiF = p_me >> 3, jjF = p_me & 7;
  const int k2b = (t & 3) * 4;          // 4 half-pairs per thread

  for (int s = 0; s < 16; ++s) {        // K-steps of 32 features
    __syncthreads();
    // stage W1 block in B-fragment order: [n][kk], kk contiguous
    {
      const uint4* g = reinterpret_cast<const uint4*>(w1T) + (t * 64 + s * 4);
      uint4* d = reinterpret_cast<uint4*>(&s_w1T[t * 32]);
#pragma unroll
      for (int q = 0; q < 4; ++q) d[q] = g[q];
    }
    // build symmetric pairwise features in A-fragment order
    {
      const int h0 = (s < 8) ? s * 32 : s * 32 - 256;
      const float* ri = &s_nodeI[iiF * 256 + h0];
      const float* rj = &s_nodeJ[jjF * 256 + h0];
#pragma unroll
      for (int u = 0; u < 4; ++u) {
        const int k2 = k2b + u, kk = k2 * 2;
        const float a0 = ri[kk], a1 = ri[kk + 1];
        const float c0 = rj[kk], c1 = rj[kk + 1];
        float v0, v1;
        if (s < 8) { v0 = fabsf(a0 - c0); v1 = fabsf(a1 - c1); }
        else       { v0 = a0 * c0;        v1 = a1 * c1; }
        const int hg = (k2 >> 2) & 1;
        const int pp = (k2 & 3) + ((k2 >> 3) << 2);
        h2 pv; pv.x = (_Float16)v0; pv.y = (_Float16)v1;
        *reinterpret_cast<h2*>(&s_feat[(p_me * 16 + k2) * 2]) = pv;
      }
    }
    __syncthreads();
    // A fragment: this wave's M-tile (16 pairs, 32 K)
    const v16h afrag = *reinterpret_cast<const v16h*>(
        &s_feat[(mt * 16 + (lane & 15)) * 32 + (lane >> 4) * 16]);
#pragma unroll
    for (int nt = 0; nt < 8; ++nt) {
      const v16h bfrag = *reinterpret_cast<const v16h*>(
          &s_w1T[((nh * 8 + nt) * 16 + (lane & 15)) * 32 + (lane >> 4) * 16]);
      acc[nt] = __builtin_amdgcn_wmma_f32_16x16x32_f16(
          false, afrag, false, bfrag, (short)0, acc[nt], false, false);
    }
  }

  // -------- dual epilogue: (i,j) and (j,i) orientations --------------------
  __syncthreads();
  {
    // stage pre1/pre2 for both i-rows and j-rows: 4 regions x 8x256 f32
    const int r2 = t >> 6;              // region
    const int w6 = t & 63;
    const int row = w6 >> 3, cb = (w6 & 7) * 32;
    const float* src;
    float* dst;
    if      (r2 == 0) { src = pre1 + ((b * NDIM + i0 + row) * HID + cb); dst = &s_p1I[row * 256 + cb]; }
    else if (r2 == 1) { src = pre2 + ((b * NDIM + j0 + row) * HID + cb); dst = &s_p2J[row * 256 + cb]; }
    else if (r2 == 2) { src = pre1 + ((b * NDIM + j0 + row) * HID + cb); dst = &s_p1J[row * 256 + cb]; }
    else              { src = pre2 + ((b * NDIM + i0 + row) * HID + cb); dst = &s_p2I[row * 256 + cb]; }
    const float4* g = reinterpret_cast<const float4*>(src);
    float4* d = reinterpret_cast<float4*>(dst);
#pragma unroll
    for (int q = 0; q < 8; ++q) d[q] = g[q];
  }
  __syncthreads();

  const int c    = lane & 15;           // N column within tile
  const int mAdd = (lane >> 4) * 8;     // C/D layout: lanes>=16 -> M+8
#pragma unroll
  for (int r = 0; r < 8; ++r) {
    const int pl = mt * 16 + r + mAdd;  // pair index 0..63
    const int ii = pl >> 3, jj = pl & 7;
    const float cF = s_clvF[pl], sF = s_sgvF[pl];
    const float cR = s_clvR[pl], sR = s_sgvR[pl];
    float partF = 0.0f, partR = 0.0f;
#pragma unroll
    for (int nt = 0; nt < 8; ++nt) {
      const int k = nh * 128 + nt * 16 + c;
      const float S   = acc[nt][r];
      const float e   = s_w1e[k], f = s_w1f[k], w = s_w2[k];
      float hF = S + s_p1I[ii * 256 + k] + s_p2J[jj * 256 + k];
      hF = fmaf(cF, e, hF); hF = fmaf(sF, f, hF);
      partF = fmaf(fmaxf(hF, 0.0f), w, partF);
      float hR = S + s_p1J[jj * 256 + k] + s_p2I[ii * 256 + k];
      hR = fmaf(cR, e, hR); hR = fmaf(sR, f, hR);
      partR = fmaf(fmaxf(hR, 0.0f), w, partR);
    }
    s_red[(pl * 2 + 0) * 32 + nh * 16 + c] = partF;
    s_red[(pl * 2 + 1) * 32 + nh * 16 + c] = partR;
  }
  __syncthreads();
  if (t < 128) {
    const int p = t >> 1, orient = t & 1;
    float sum = b2v;
#pragma unroll
    for (int c2 = 0; c2 < 32; ++c2) sum += s_red[(p * 2 + orient) * 32 + c2];
    const int ii = p >> 3, jj = p & 7;
    const int i = i0 + ii, j = j0 + jj;
    if (orient == 0) raw[(b * NDIM + i) * NDIM + j] = sum;
    else             raw[(b * NDIM + j) * NDIM + i] = sum;
  }
}

// ---------------------------------------------------------------------------
// Kernel 3: symmetrize + diagonal mask
// ---------------------------------------------------------------------------
__global__ __launch_bounds__(256)
void symmetrize_kernel(const float* __restrict__ raw, float* __restrict__ out) {
  const int idx = blockIdx.x * 256 + threadIdx.x;   // 0 .. 262143
  const int b = idx >> 16;
  const int r = idx & 0xFFFF;
  const int i = r >> 8, j = r & 255;
  float v;
  if (i == j) {
    v = -1e9f;
  } else {
    v = 0.5f * (raw[(b * NDIM + i) * NDIM + j] + raw[(b * NDIM + j) * NDIM + i]);
  }
  out[idx] = v;
}

// ---------------------------------------------------------------------------
extern "C" void kernel_launch(void* const* d_in, const int* in_sizes, int n_in,
                              void* d_out, int out_size, void* d_ws, size_t ws_size,
                              hipStream_t stream) {
  const float* node    = (const float*)d_in[0];  // [4,256,256]
  const float* logitsI = (const float*)d_in[1];  // [4,256,256]
  const float* W1      = (const float*)d_in[2];  // [1026,256]
  const float* b1      = (const float*)d_in[3];  // [256]
  const float* W2      = (const float*)d_in[4];  // [256,1]
  const float* b2      = (const float*)d_in[5];  // [1]
  float* out = (float*)d_out;

  char* ws = (char*)d_ws;
  float*    pre1 = (float*)(ws);                             // 1 MB
  float*    pre2 = (float*)(ws + (1u << 20));                // 1 MB
  _Float16* w1T  = (_Float16*)(ws + (2u << 20));             // 256 KB [256][512]
  float*    raw  = (float*)(ws + (2u << 20) + (256u << 10)); // 1 MB

  precompute_kernel<<<BDIM * NDIM, 256, 0, stream>>>(node, W1, b1, pre1, pre2);
  w1t_kernel<<<512, 256, 0, stream>>>(W1, w1T);
  dim3 grid(NTRI, BDIM);   // 528 upper-triangular tile pairs x 4 batches
  edge_mlp_main<<<grid, 256, 0, stream>>>(node, logitsI, W1, W2, b2,
                                          pre1, pre2, w1T, raw);
  symmetrize_kernel<<<(BDIM * NDIM * NDIM) / 256, 256, 0, stream>>>(raw, out);
}
